// DualMemoryElmanOptimized_2147483648051
// MI455X (gfx1250) — matrix operand order repaired and checked by reading, not compiled, not run on hardware
//
#include <hip/hip_runtime.h>
#include <hip/hip_bf16.h>

// ---------------------------------------------------------------------------
// DualMemoryElman for MI455X (gfx1250): bf16 WMMA everywhere matrix-shaped,
// f32 accumulate. Streaming GEMMs are HBM-bound (~300MB @ 23.3TB/s); the
// sequential scan keeps h_tape slices resident in LDS, weights in L2, and
// runs as a persistent 64-WG kernel with device-wide barriers.
// GEMM staging uses the CDNA5 async global->LDS path (ASYNCcnt).
// ---------------------------------------------------------------------------

typedef __attribute__((ext_vector_type(16))) __bf16 v16bf;
typedef __attribute__((ext_vector_type(8)))  float  v8f;

union FragAB {
  v16bf v;
  uint4 q[2];
};

__device__ __forceinline__ unsigned short f32_to_bf16(float f) {
  union { float f; unsigned u; } x; x.f = f;
  unsigned u = x.u;
  unsigned r = u + 0x7FFFu + ((u >> 16) & 1u);   // round-to-nearest-even
  return (unsigned short)(r >> 16);
}
__device__ __forceinline__ float bf16_to_f32(unsigned short h) {
  union { unsigned u; float f; } x; x.u = ((unsigned)h) << 16;
  return x.f;
}

// CDNA5 async global->LDS copy, 16B per lane (tracked by ASYNCcnt).
// lds_off: wave-relative LDS byte address (low 32 bits of a __shared__ ptr).
__device__ __forceinline__ void async_g2l_b128(unsigned lds_off,
                                               const void* gaddr) {
  asm volatile("global_load_async_to_lds_b128 %0, %1, off"
               :: "v"(lds_off), "v"(gaddr)
               : "memory");
}
__device__ __forceinline__ void wait_async0() {
  asm volatile("s_wait_asynccnt 0x0" ::: "memory");
}

// ---------------------------------------------------------------------------
// f32 -> bf16 conversion
// ---------------------------------------------------------------------------
__global__ void cvt_bf16_kernel(const float* __restrict__ src,
                                unsigned short* __restrict__ dst, int n) {
  for (int i = blockIdx.x * blockDim.x + threadIdx.x; i < n;
       i += gridDim.x * blockDim.x)
    dst[i] = f32_to_bf16(src[i]);
}

// ---------------------------------------------------------------------------
// Blocked bf16 WMMA GEMM:  O[M,N] = A[M,K] @ W[N,K]^T   (i.e. y = x @ W.T)
// 64x64 block, 4 waves (2x2), each wave 32x32 = 2x2 tiles of 16x16, K step 64
// (8 WMMAs per staging round). Panels staged with async global->LDS copies.
// mode 0: store f32; mode 1: store bf16; mode 3: silu -> bf16.
// ---------------------------------------------------------------------------
#define GEMM_BM 64
#define GEMM_BN 64
#define GEMM_BK 64

__global__ void __launch_bounds__(128)
gemm_bf16_wmma(const unsigned short* __restrict__ A,
               const unsigned short* __restrict__ W,
               float* __restrict__ outF,
               unsigned short* __restrict__ outH,
               int M, int N, int K, int mode) {
  __shared__ __align__(16) unsigned short As[GEMM_BM * GEMM_BK];
  __shared__ __align__(16) unsigned short Bs[GEMM_BN * GEMM_BK];

  const int m0 = blockIdx.x * GEMM_BM;
  const int n0 = blockIdx.y * GEMM_BN;
  const int tid  = threadIdx.x;
  const int wave = tid >> 5, lane = tid & 31;
  const int wm = wave >> 1, wn = wave & 1;   // 2x2 wave grid
  const int lm = lane & 15, hi = lane >> 4;

  const unsigned as_base = (unsigned)(size_t)(const void*)&As[0];
  const unsigned bs_base = (unsigned)(size_t)(const void*)&Bs[0];

  v8f acc[2][2] = {};

  for (int k0 = 0; k0 < K; k0 += GEMM_BK) {
    __syncthreads();
    // Stage 64x64 bf16 panels: 4x 16B async chunks/thread/panel.
#pragma unroll
    for (int i = 0; i < 4; ++i) {
      int c   = tid + i * 128;        // 0..511
      int row = c >> 3;
      int kc  = (c & 7) * 8;
      async_g2l_b128(as_base + (unsigned)(row * GEMM_BK + kc) * 2u,
                     &A[(size_t)(m0 + row) * K + k0 + kc]);
      async_g2l_b128(bs_base + (unsigned)(row * GEMM_BK + kc) * 2u,
                     &W[(size_t)(n0 + row) * K + k0 + kc]);
    }
    wait_async0();
    __syncthreads();

#pragma unroll
    for (int ks = 0; ks < GEMM_BK; ks += 32) {
      // A fragment (16x32 bf16): lane m=lm, hi half -> K chunks {hi*8,16+hi*8}
      FragAB fa[2], fb[2];
#pragma unroll
      for (int mt = 0; mt < 2; ++mt) {
        int mrow = wm * 32 + mt * 16 + lm;
        fa[mt].q[0] =
            *reinterpret_cast<const uint4*>(&As[mrow * GEMM_BK + ks + hi * 8]);
        fa[mt].q[1] = *reinterpret_cast<const uint4*>(
            &As[mrow * GEMM_BK + ks + 16 + hi * 8]);
      }
      // B fragment (32x16 bf16): lane n=lm, hi half -> K chunks {hi*8,16+hi*8}
#pragma unroll
      for (int nt = 0; nt < 2; ++nt) {
        int nrow = wn * 32 + nt * 16 + lm;
        fb[nt].q[0] =
            *reinterpret_cast<const uint4*>(&Bs[nrow * GEMM_BK + ks + hi * 8]);
        fb[nt].q[1] = *reinterpret_cast<const uint4*>(
            &Bs[nrow * GEMM_BK + ks + 16 + hi * 8]);
      }
#pragma unroll
      for (int mt = 0; mt < 2; ++mt)
#pragma unroll
        for (int nt = 0; nt < 2; ++nt)
          acc[mt][nt] = __builtin_amdgcn_wmma_f32_16x16x32_bf16(
              false, fa[mt].v, false, fb[nt].v, (short)0, acc[mt][nt], false,
              false);
    }
  }

  // Epilogue per documented C/D layout: lane<16 -> rows 0..7, lane>=16 -> 8..15
#pragma unroll
  for (int mt = 0; mt < 2; ++mt) {
#pragma unroll
    for (int nt = 0; nt < 2; ++nt) {
      int col = n0 + wn * 32 + nt * 16 + lm;
#pragma unroll
      for (int r = 0; r < 8; ++r) {
        int row = m0 + wm * 32 + mt * 16 + hi * 8 + r;
        float v = acc[mt][nt][r];
        size_t idx = (size_t)row * N + col;
        if (mode == 0) {
          outF[idx] = v;
        } else if (mode == 1) {
          outH[idx] = f32_to_bf16(v);
        } else {  // silu -> bf16
          float s = v / (1.f + expf(-v));
          outH[idx] = f32_to_bf16(s);
        }
      }
    }
  }
}

// ---------------------------------------------------------------------------
// Persistent recurrent scan.
// Grid: 64 WGs x 256 threads (8 waves). WG g owns output/d slice [16g,16g+16).
// Its h_tape slice [16,64,16] f32 (64KB) lives in dynamic LDS for all steps.
// Per step: (B) Rh slice via 8-wave K-split WMMA + dot1 partials -> barrier
//           (D) reduce dots, softmax, read, h_new = tanh(...)     -> barrier
//           (G) write_val slice via WMMA + dot2 partials          -> barrier
//           softmax(wa), tape slice update in LDS.
// ---------------------------------------------------------------------------
#define SB 16
#define SD 1024
#define SNS 64
#define ST 1024
#define SCAN_WGS 64
#define SCAN_THREADS 256
#define SCALE_F 0.03125f   // 1/sqrt(1024)

__device__ __forceinline__ void grid_barrier(int* cnt, volatile int* gen) {
  __syncthreads();
  if (threadIdx.x == 0) {
    __threadfence();
    int g = *gen;
    if (atomicAdd(cnt, 1) == SCAN_WGS - 1) {
      *cnt = 0;
      __threadfence();
      atomicAdd((int*)gen, 1);
    } else {
      while (*gen == g) { __builtin_amdgcn_s_sleep(1); }
      __threadfence();
    }
  }
  __syncthreads();
}

// One 16x16 output slice of [16,1024]x[1024,1024]^T, K split across 8 waves
// (4 WMMAs each), partials dumped to LDS for cross-wave reduction.
__device__ __forceinline__ void slice_gemm_partial(
    const unsigned short* __restrict__ hwbf,
    const unsigned short* __restrict__ Wbf,
    int g, int wave, int lm, int hi, float (*partial)[256]) {
  v8f acc = {};
  const int kbase = wave * 128;
#pragma unroll
  for (int kk = 0; kk < 128; kk += 32) {
    const int k0 = kbase + kk;
    FragAB fa, fb;
    fa.q[0] = *reinterpret_cast<const uint4*>(&hwbf[lm * SD + k0 + hi * 8]);
    fa.q[1] = *reinterpret_cast<const uint4*>(&hwbf[lm * SD + k0 + 16 + hi * 8]);
    const unsigned short* wr = &Wbf[(size_t)(g * 16 + lm) * SD + k0 + hi * 8];
    fb.q[0] = *reinterpret_cast<const uint4*>(wr);
    fb.q[1] = *reinterpret_cast<const uint4*>(wr + 16);
    acc = __builtin_amdgcn_wmma_f32_16x16x32_bf16(false, fa.v, false, fb.v,
                                                  (short)0, acc, false, false);
  }
#pragma unroll
  for (int r = 0; r < 8; ++r)
    partial[wave][(hi * 8 + r) * 16 + lm] = acc[r];
}

__global__ void __launch_bounds__(SCAN_THREADS)
scan_kernel(const unsigned short* __restrict__ Whbf,
            const unsigned short* __restrict__ Wwbf,
            const float* __restrict__ wx,
            const float* __restrict__ b_h,
            float* __restrict__ hw_f32,
            unsigned short* __restrict__ hw_bf,
            unsigned short* __restrict__ h_all_bf,
            float* __restrict__ dots1,
            float* __restrict__ dots2,
            int* bar_cnt, volatile int* bar_gen,
            float* __restrict__ out_tape,
            float* __restrict__ out_hwork) {
  extern __shared__ float tape[];            // [16][64][16] f32 = 64KB
  __shared__ float partial[8][256];
  __shared__ float hw_sl[256];               // h_work slice [b][j]
  __shared__ float rh_sl[256];
  __shared__ float wv_sl[256];
  __shared__ float dots_l[SB * SNS];         // dot / attn scratch [b][n]

  const int g    = blockIdx.x;
  const int tid  = threadIdx.x;
  const int wave = tid >> 5, lane = tid & 31;
  const int lm = lane & 15, hi = lane >> 4;
  const int tb = tid >> 4, tj = tid & 15;    // (b, j) owned by this thread

  for (int i = tid; i < SB * SNS * 16; i += SCAN_THREADS) tape[i] = 0.f;
  hw_sl[tid] = 0.f;
  __syncthreads();

  for (int t = 0; t < ST; ++t) {
    // ---- Phase B: Rh = h_work @ W_h.T (slice), dot1 partials -------------
    slice_gemm_partial(hw_bf, Whbf, g, wave, lm, hi, partial);
    __syncthreads();
    {
      float s = 0.f;
#pragma unroll
      for (int w = 0; w < 8; ++w) s += partial[w][tid];
      rh_sl[tid] = s;
    }
#pragma unroll
    for (int i = 0; i < 4; ++i) {
      int p = tid * 4 + i;                   // p = b*64 + n
      int b = p >> 6, n = p & 63;
      float s = 0.f;
#pragma unroll
      for (int j = 0; j < 16; ++j)
        s += hw_sl[b * 16 + j] * tape[(b * SNS + n) * 16 + j];
      dots1[g * (SB * SNS) + p] = s;
    }
    // prefetch next step's wx slice while we wait at the barrier
    if (t + 1 < ST)
      __builtin_prefetch(&wx[((size_t)tb * ST + (t + 1)) * SD + g * 16 + tj],
                         0, 0);
    __threadfence();
    grid_barrier(bar_cnt, bar_gen);

    // ---- Phase D: read-attention softmax, read, h_new --------------------
    for (int i = 0; i < 4; ++i) {
      int p = tid * 4 + i;
      float s = 0.f;
      for (int gg = 0; gg < SCAN_WGS; ++gg) s += dots1[gg * (SB * SNS) + p];
      dots_l[p] = s;
    }
    __syncthreads();
    if (tid < SB) {
      int b = tid;
      float mx = -1e30f;
      for (int n = 0; n < SNS; ++n) mx = fmaxf(mx, dots_l[b * SNS + n] * SCALE_F);
      float sum = 0.f;
      for (int n = 0; n < SNS; ++n) {
        float e = expf(dots_l[b * SNS + n] * SCALE_F - mx);
        dots_l[b * SNS + n] = e; sum += e;
      }
      float inv = 1.f / sum;
      for (int n = 0; n < SNS; ++n) dots_l[b * SNS + n] *= inv;
    }
    __syncthreads();
    {
      float rd = 0.f;
      for (int n = 0; n < SNS; ++n)
        rd += dots_l[tb * SNS + n] * tape[(tb * SNS + n) * 16 + tj];
      int col = g * 16 + tj;
      float h = tanhf(rh_sl[tid] + wx[((size_t)tb * ST + t) * SD + col] + rd +
                      b_h[col]);
      hw_sl[tid] = h;
      hw_f32[tb * SD + col] = h;
      unsigned short hb = f32_to_bf16(h);
      hw_bf[tb * SD + col] = hb;
      h_all_bf[((size_t)tb * ST + t) * SD + col] = hb;
      if (t == ST - 1) out_hwork[tb * SD + col] = h;
    }
    __threadfence();
    grid_barrier(bar_cnt, bar_gen);

    // ---- Phase G: write_val = h_new @ W_write.T (slice), dot2 partials ---
    slice_gemm_partial(hw_bf, Wwbf, g, wave, lm, hi, partial);
    __syncthreads();
    {
      float s = 0.f;
#pragma unroll
      for (int w = 0; w < 8; ++w) s += partial[w][tid];
      wv_sl[tid] = s;
    }
#pragma unroll
    for (int i = 0; i < 4; ++i) {
      int p = tid * 4 + i;
      int b = p >> 6, n = p & 63;
      float s = 0.f;
#pragma unroll
      for (int j = 0; j < 16; ++j)
        s += hw_sl[b * 16 + j] * tape[(b * SNS + n) * 16 + j];
      dots2[g * (SB * SNS) + p] = s;
    }
    __threadfence();
    grid_barrier(bar_cnt, bar_gen);

    // ---- write-attention softmax + tape slice update ---------------------
    for (int i = 0; i < 4; ++i) {
      int p = tid * 4 + i;
      float s = 0.f;
      for (int gg = 0; gg < SCAN_WGS; ++gg) s += dots2[gg * (SB * SNS) + p];
      dots_l[p] = s;
    }
    __syncthreads();
    if (tid < SB) {
      int b = tid;
      float mx = -1e30f;
      for (int n = 0; n < SNS; ++n) mx = fmaxf(mx, dots_l[b * SNS + n] * SCALE_F);
      float sum = 0.f;
      for (int n = 0; n < SNS; ++n) {
        float e = expf(dots_l[b * SNS + n] * SCALE_F - mx);
        dots_l[b * SNS + n] = e; sum += e;
      }
      float inv = 1.f / sum;
      for (int n = 0; n < SNS; ++n) dots_l[b * SNS + n] *= inv;
    }
    __syncthreads();
    for (int idx = tid; idx < SB * SNS * 16; idx += SCAN_THREADS) {
      int j = idx & 15, n = (idx >> 4) & 63, b = idx >> 10;
      float wa = dots_l[b * SNS + n];
      float nv = (1.f - wa) * tape[idx] + wa * wv_sl[b * 16 + j];
      tape[idx] = nv;
      if (t == ST - 1)
        out_tape[(size_t)(b * SNS + n) * SD + g * 16 + j] = nv;
    }
    __syncthreads();
  }
}

// ---------------------------------------------------------------------------
// Elementwise: hg = bf16( h_all * gate )
// ---------------------------------------------------------------------------
__global__ void hg_kernel(const unsigned short* __restrict__ h,
                          const unsigned short* __restrict__ gt,
                          unsigned short* __restrict__ hg, int n) {
  for (int i = blockIdx.x * blockDim.x + threadIdx.x; i < n;
       i += gridDim.x * blockDim.x)
    hg[i] = f32_to_bf16(bf16_to_f32(h[i]) * bf16_to_f32(gt[i]));
}

// ---------------------------------------------------------------------------
// Per-launch init: zero h_work state + device barrier
// ---------------------------------------------------------------------------
__global__ void scan_init_kernel(float* hwf, unsigned short* hwb, int n,
                                 int* bar) {
  int i = blockIdx.x * blockDim.x + threadIdx.x;
  if (i < n) { hwf[i] = 0.f; hwb[i] = 0; }
  if (i == 0) { bar[0] = 0; bar[1] = 0; }
}

// ---------------------------------------------------------------------------
extern "C" void kernel_launch(void* const* d_in, const int* in_sizes, int n_in,
                              void* d_out, int out_size, void* d_ws,
                              size_t ws_size, hipStream_t stream) {
  const float* x        = (const float*)d_in[0];  // [16,1024,1024]
  const float* in_proj  = (const float*)d_in[1];  // [1024,1024]
  const float* out_proj = (const float*)d_in[2];
  const float* gate_proj= (const float*)d_in[3];
  const float* W_x      = (const float*)d_in[4];
  const float* W_h      = (const float*)d_in[5];
  const float* W_write  = (const float*)d_in[6];
  const float* b_h      = (const float*)d_in[7];

  const int BT = SB * ST;                 // 16384 rows
  const int D  = SD;                      // 1024
  const size_t NTOK = (size_t)BT * D;     // 16,777,216
  const size_t NW   = (size_t)D * D;      // 1,048,576

  // workspace carve (256B aligned)
  size_t off = 0;
  auto carve = [&](size_t bytes) -> char* {
    char* p = (char*)d_ws + off;
    off += (bytes + 255) & ~(size_t)255;
    return p;
  };
  unsigned short* x_bf     = (unsigned short*)carve(NTOK * 2);
  unsigned short* w_in     = (unsigned short*)carve(NW * 2);
  unsigned short* w_wx     = (unsigned short*)carve(NW * 2);
  unsigned short* w_gate   = (unsigned short*)carve(NW * 2);
  unsigned short* w_out    = (unsigned short*)carve(NW * 2);
  unsigned short* w_wh     = (unsigned short*)carve(NW * 2);
  unsigned short* w_ww     = (unsigned short*)carve(NW * 2);
  unsigned short* xproj_bf = (unsigned short*)carve(NTOK * 2);
  float*          wx_f32   = (float*)carve(NTOK * 4);
  unsigned short* gate_bf  = (unsigned short*)carve(NTOK * 2);
  unsigned short* h_all_bf = (unsigned short*)carve(NTOK * 2);
  unsigned short* hg_bf    = (unsigned short*)carve(NTOK * 2);
  float*          hw_f32   = (float*)carve((size_t)SB * D * 4);
  unsigned short* hw_bf    = (unsigned short*)carve((size_t)SB * D * 2);
  float*          dots1    = (float*)carve((size_t)SCAN_WGS * SB * SNS * 4);
  float*          dots2    = (float*)carve((size_t)SCAN_WGS * SB * SNS * 4);
  int*            barrier  = (int*)carve(2 * sizeof(int));
  (void)ws_size; (void)in_sizes; (void)n_in; (void)out_size;

  float* out_f     = (float*)d_out;                 // output [16,1024,1024]
  float* out_tape  = out_f + NTOK;                  // h_tape_f [16,64,1024]
  float* out_hwork = out_tape + (size_t)SB * SNS * D; // h_work_f [16,1024]

  // 1) f32 -> bf16 conversions
  cvt_bf16_kernel<<<4096, 256, 0, stream>>>(x, x_bf, (int)NTOK);
  cvt_bf16_kernel<<<1024, 256, 0, stream>>>(in_proj,  w_in,  (int)NW);
  cvt_bf16_kernel<<<1024, 256, 0, stream>>>(W_x,      w_wx,  (int)NW);
  cvt_bf16_kernel<<<1024, 256, 0, stream>>>(gate_proj,w_gate,(int)NW);
  cvt_bf16_kernel<<<1024, 256, 0, stream>>>(out_proj, w_out, (int)NW);
  cvt_bf16_kernel<<<1024, 256, 0, stream>>>(W_h,      w_wh,  (int)NW);
  cvt_bf16_kernel<<<1024, 256, 0, stream>>>(W_write,  w_ww,  (int)NW);
  scan_init_kernel<<<64, 256, 0, stream>>>(hw_f32, hw_bf, SB * D, barrier);

  dim3 gblk(128);
  dim3 ggrid(BT / GEMM_BM, D / GEMM_BN);

  // 2) x_proj = x @ in_proj.T   (bf16 out)
  gemm_bf16_wmma<<<ggrid, gblk, 0, stream>>>(x_bf, w_in, nullptr, xproj_bf,
                                             BT, D, D, 1);
  // 3) wx = x_proj @ W_x.T      (f32 out)
  gemm_bf16_wmma<<<ggrid, gblk, 0, stream>>>(xproj_bf, w_wx, wx_f32, nullptr,
                                             BT, D, D, 0);
  // 4) gate = silu(x @ gate_proj.T)  (bf16 out)
  gemm_bf16_wmma<<<ggrid, gblk, 0, stream>>>(x_bf, w_gate, nullptr, gate_bf,
                                             BT, D, D, 3);

  // 5) persistent recurrent scan (64 WGs, 64KB dynamic LDS tape slice each)
  scan_kernel<<<SCAN_WGS, SCAN_THREADS, SB * SNS * 16 * sizeof(float), stream>>>(
      w_wh, w_ww, wx_f32, b_h, hw_f32, hw_bf, h_all_bf, dots1, dots2,
      barrier, (volatile int*)(barrier + 1), out_tape, out_hwork);

  // 6) hg = h_all * gate
  hg_kernel<<<4096, 256, 0, stream>>>(h_all_bf, gate_bf, hg_bf, (int)NTOK);

  // 7) output = hg @ out_proj.T  (f32 into d_out)
  gemm_bf16_wmma<<<ggrid, gblk, 0, stream>>>(hg_bf, w_out, out_f, nullptr,
                                             BT, D, D, 0);
}
